// AttModel_24678882083678
// MI455X (gfx1250) — matrix operand order repaired and checked by reading, not compiled, not run on hardware
//
#include <hip/hip_runtime.h>
#include <stdint.h>

#define N_BOX 1024
#define H_IMG 4096
#define W_IMG 4096
#define IOU_T 0.5f
#define NEG_INF -1e30f

typedef __attribute__((ext_vector_type(16))) __bf16     v16bf;
typedef __attribute__((ext_vector_type(8)))  float      v8f;
typedef __attribute__((ext_vector_type(8)))  unsigned   v8u;

static __device__ __forceinline__ unsigned f32_to_bf16_rne(float f) {
  unsigned u = __float_as_uint(f);
  return (u + 0x7FFFu + ((u >> 16) & 1u)) >> 16;
}

// 8-bit indicator mask for k in [k0, k0+8) against [y1, y2)
static __device__ __forceinline__ unsigned runmask8(int y1, int y2, int k0) {
  int lo = min(max(y1 - k0, 0), 8);
  int hi = min(max(y2 - k0, 0), 8);
  return (1u << hi) - (1u << lo);
}

// ---------------------------------------------------------------------------
// Kernel 1: S = R x P via WMMA bf16, masked by x-range, partial box sums.
// Grid: 256 blocks (one 16-wide x slice each), 1024 threads (32 waves).
// Each wave owns 2 m-tiles (16 boxes each) => 64 tiles cover all 1024 boxes,
// so the 64MB probs array streams from HBM exactly once for the whole kernel
// (and lives in the 192MB L2 afterwards). A-matrix (row indicators) costs
// zero bytes: fragments are synthesized from (y1,y2).
//   - wave-uniform zero-skip: k-step outside the union of the wave's 32 box
//     y-spans -> skip B loads + WMMAs entirely (~20% of steps)
//   - wave-uniform fast path: every lane's 8-runs uniformly 0/1 (~75%) ->
//     fragment from two cndmask constants
//   - else: 256-entry LDS LUT expands mask8 -> 4 dwords, 1 ds_load_b128/run
// ---------------------------------------------------------------------------
__global__ __launch_bounds__(1024) void boxsum_gemm(
    const float* __restrict__ bbox, const float* __restrict__ probs,
    float* __restrict__ partial) {
  __shared__ unsigned short Bs[1024];  // 2 k-steps of B tile [32k x 16n], frag-swizzled
  __shared__ uint4 lutA[256];          // mask8 -> 4 dwords of packed bf16 {0,1}
  __shared__ int bbx1[N_BOX], bby1[N_BOX], bbx2[N_BOX], bby2[N_BOX];

  const int t = threadIdx.x;
  const int xbase = blockIdx.x * 16;

  {  // integer box bounds (floor + clip), one box per thread
    float x1f = bbox[t * 4 + 0], y1f = bbox[t * 4 + 1];
    float x2f = bbox[t * 4 + 2], y2f = bbox[t * 4 + 3];
    bbx1[t] = min(max((int)floorf(x1f), 0), W_IMG);
    bby1[t] = min(max((int)floorf(y1f), 0), H_IMG);
    bbx2[t] = min(max((int)floorf(x2f), 0), W_IMG);
    bby2[t] = min(max((int)floorf(y2f), 0), H_IMG);
  }
  if (t < 256) {  // build expansion LUT
    uint4 e;
    unsigned m8 = (unsigned)t;
    e.x = ((m8 & 0x01u) ? 0x00003F80u : 0u) | ((m8 & 0x02u) ? 0x3F800000u : 0u);
    e.y = ((m8 & 0x04u) ? 0x00003F80u : 0u) | ((m8 & 0x08u) ? 0x3F800000u : 0u);
    e.z = ((m8 & 0x10u) ? 0x00003F80u : 0u) | ((m8 & 0x20u) ? 0x3F800000u : 0u);
    e.w = ((m8 & 0x40u) ? 0x00003F80u : 0u) | ((m8 & 0x80u) ? 0x3F800000u : 0u);
    lutA[t] = e;
  }
  __syncthreads();

  const int lane = t & 31;
  const int wv   = t >> 5;
  const int l16  = lane & 15;
  const int ko   = (lane < 16) ? 0 : 8;  // A-frag K offset per half-wave

  const int mb0 = 16 * (2 * wv + 0);
  const int mb1 = 16 * (2 * wv + 1);
  const int y1t0 = bby1[mb0 + l16], y2t0 = bby2[mb0 + l16];
  const int y1t1 = bby1[mb1 + l16], y2t1 = bby2[mb1 + l16];

  v8f acc0 = {};
  v8f acc1 = {};

  // staging assignment: thread t moves one f32 -> bf16 element per 64-row chunk
  const int sk  = t >> 4;   // row in chunk 0..63
  const int sn  = t & 15;   // col 0..15
  const int k32 = sk & 31;
  const int sOff = (sk >> 5) * 512 + (((k32 >> 3) & 1) * 16 + sn) * 16 + ((k32 >> 4) << 3) + (k32 & 7);
  const float* gsrc = probs + (size_t)sk * W_IMG + xbase + sn;

  for (int chunk = 0; chunk < 64; ++chunk) {
    float pv = gsrc[(size_t)chunk * 64 * W_IMG];
    if (chunk + 1 < 64)
      __builtin_prefetch(gsrc + (size_t)(chunk + 1) * 64 * W_IMG, 0, 0);
    Bs[sOff] = (unsigned short)f32_to_bf16_rne(pv);
    __syncthreads();

#pragma unroll
    for (int sub = 0; sub < 2; ++sub) {
      const int kb = chunk * 64 + sub * 32;

      // per-lane 8-run masks for both tiles
      unsigned ma0 = runmask8(y1t0, y2t0, kb + ko);
      unsigned mb0m = runmask8(y1t0, y2t0, kb + ko + 16);
      unsigned ma1 = runmask8(y1t1, y2t1, kb + ko);
      unsigned mb1m = runmask8(y1t1, y2t1, kb + ko + 16);

      // wave-uniform skip: this k-step misses all 32 boxes of both tiles
      unsigned anyA = ma0 | mb0m | ma1 | mb1m;
      if (__all((int)(anyA == 0u))) continue;

      // B fragment: contiguous 32B per lane (two ds_load_b128)
      const unsigned* bp = (const unsigned*)(Bs + sub * 512 + lane * 16);
      union { v8u u; v16bf b; } B;
#pragma unroll
      for (int q = 0; q < 8; ++q) B.u[q] = bp[q];

      union { v8u u; v16bf b; } A0, A1;
      bool fast = ((ma0 == 0u) | (ma0 == 0xFFu)) &
                  ((mb0m == 0u) | (mb0m == 0xFFu)) &
                  ((ma1 == 0u) | (ma1 == 0xFFu)) &
                  ((mb1m == 0u) | (mb1m == 0xFFu));
      if (__all((int)fast)) {
        // uniform fast path: every lane fully inside/outside -> constants
        unsigned wa0 = (ma0 & 1u) ? 0x3F803F80u : 0u;
        unsigned wb0 = (mb0m & 1u) ? 0x3F803F80u : 0u;
        unsigned wa1 = (ma1 & 1u) ? 0x3F803F80u : 0u;
        unsigned wb1 = (mb1m & 1u) ? 0x3F803F80u : 0u;
        A0.u[0] = wa0; A0.u[1] = wa0; A0.u[2] = wa0; A0.u[3] = wa0;
        A0.u[4] = wb0; A0.u[5] = wb0; A0.u[6] = wb0; A0.u[7] = wb0;
        A1.u[0] = wa1; A1.u[1] = wa1; A1.u[2] = wa1; A1.u[3] = wa1;
        A1.u[4] = wb1; A1.u[5] = wb1; A1.u[6] = wb1; A1.u[7] = wb1;
      } else {
        // LUT expansion: one ds_load_b128 per 8-run
        uint4 ra0 = lutA[ma0], rb0 = lutA[mb0m];
        uint4 ra1 = lutA[ma1], rb1 = lutA[mb1m];
        A0.u[0] = ra0.x; A0.u[1] = ra0.y; A0.u[2] = ra0.z; A0.u[3] = ra0.w;
        A0.u[4] = rb0.x; A0.u[5] = rb0.y; A0.u[6] = rb0.z; A0.u[7] = rb0.w;
        A1.u[0] = ra1.x; A1.u[1] = ra1.y; A1.u[2] = ra1.z; A1.u[3] = ra1.w;
        A1.u[4] = rb1.x; A1.u[5] = rb1.y; A1.u[6] = rb1.z; A1.u[7] = rb1.w;
      }

      acc0 = __builtin_amdgcn_wmma_f32_16x16x32_bf16(false, A0.b, false, B.b,
                                                     (short)0, acc0, false, false);
      acc1 = __builtin_amdgcn_wmma_f32_16x16x32_bf16(false, A1.b, false, B.b,
                                                     (short)0, acc1, false, false);
    }
    __syncthreads();
  }

  // mask by x-range, reduce 16 lanes per row, emit partial sums
  const int nx = xbase + l16;
  const int mhi = (lane >= 16) ? 8 : 0;
#pragma unroll
  for (int v = 0; v < 8; ++v) {
    int m0 = mb0 + v + mhi;
    float s0 = ((nx >= bbx1[m0]) && (nx < bbx2[m0])) ? acc0[v] : 0.0f;
    s0 += __shfl_xor(s0, 1, 16);
    s0 += __shfl_xor(s0, 2, 16);
    s0 += __shfl_xor(s0, 4, 16);
    s0 += __shfl_xor(s0, 8, 16);
    if (l16 == v) partial[blockIdx.x * N_BOX + m0] = s0;

    int m1 = mb1 + v + mhi;
    float s1 = ((nx >= bbx1[m1]) && (nx < bbx2[m1])) ? acc1[v] : 0.0f;
    s1 += __shfl_xor(s1, 1, 16);
    s1 += __shfl_xor(s1, 2, 16);
    s1 += __shfl_xor(s1, 4, 16);
    s1 += __shfl_xor(s1, 8, 16);
    if (l16 == v) partial[blockIdx.x * N_BOX + m1] = s1;
  }
}

// ---------------------------------------------------------------------------
// Kernel 2: deterministic fixed-order reduction of partials -> scores
// ---------------------------------------------------------------------------
__global__ __launch_bounds__(256) void finalize_scores(
    const float* __restrict__ bbox, const float* __restrict__ obj,
    const float* __restrict__ partial, float* __restrict__ score) {
  int b = blockIdx.x * blockDim.x + threadIdx.x;
  if (b >= N_BOX) return;
  float s = 0.0f;
  for (int c = 0; c < 256; ++c) s += partial[c * N_BOX + b];
  int x1 = min(max((int)floorf(bbox[b * 4 + 0]), 0), W_IMG);
  int y1 = min(max((int)floorf(bbox[b * 4 + 1]), 0), H_IMG);
  int x2 = min(max((int)floorf(bbox[b * 4 + 2]), 0), W_IMG);
  int y2 = min(max((int)floorf(bbox[b * 4 + 3]), 0), H_IMG);
  int cnt = max((y2 - y1) * (x2 - x1), 1);
  score[b] = 0.5f * (obj[b] + s / (float)cnt);
}

// ---------------------------------------------------------------------------
// Kernel 3: single-block bitonic sort, (score desc, index asc) == stable
// argsort(-scores); gathers sorted bbox.
// ---------------------------------------------------------------------------
__global__ __launch_bounds__(1024) void sort_scores(
    const float* __restrict__ bbox, const float* __restrict__ score,
    float* __restrict__ scores_s, float* __restrict__ bbox_s,
    int* __restrict__ order) {
  __shared__ float sk_[N_BOX];
  __shared__ int   si_[N_BOX];
  const int t = threadIdx.x;
  sk_[t] = score[t];
  si_[t] = t;
  __syncthreads();
  for (int k = 2; k <= N_BOX; k <<= 1) {
    for (int j = k >> 1; j > 0; j >>= 1) {
      int ixj = t ^ j;
      if (ixj > t) {
        float a = sk_[t], b = sk_[ixj];
        int ia = si_[t], ib = si_[ixj];
        bool before = (a > b) || (a == b && ia < ib);  // a precedes b (desc)
        bool doswap = ((t & k) == 0) ? (!before) : before;
        if (doswap) { sk_[t] = b; si_[t] = ib; sk_[ixj] = a; si_[ixj] = ia; }
      }
      __syncthreads();
    }
  }
  scores_s[t] = sk_[t];
  int o = si_[t];
  order[t] = o;
  bbox_s[t * 4 + 0] = bbox[o * 4 + 0];
  bbox_s[t * 4 + 1] = bbox[o * 4 + 1];
  bbox_s[t * 4 + 2] = bbox[o * 4 + 2];
  bbox_s[t * 4 + 3] = bbox[o * 4 + 3];
}

// ---------------------------------------------------------------------------
// Kernel 4: bit-packed IoU>T matrix via wave32 ballot (one wave per row)
// ---------------------------------------------------------------------------
__global__ __launch_bounds__(256) void iou_bits(
    const float* __restrict__ bbox_s, unsigned* __restrict__ iou) {
  const int row  = (blockIdx.x * blockDim.x + threadIdx.x) >> 5;
  const int lane = threadIdx.x & 31;
  float ax1 = bbox_s[row * 4 + 0], ay1 = bbox_s[row * 4 + 1];
  float ax2 = bbox_s[row * 4 + 2], ay2 = bbox_s[row * 4 + 3];
  float areaA = (ax2 - ax1) * (ay2 - ay1);
  for (int c = 0; c < 32; ++c) {
    int j = c * 32 + lane;
    float bx1 = bbox_s[j * 4 + 0], by1 = bbox_s[j * 4 + 1];
    float bx2 = bbox_s[j * 4 + 2], by2 = bbox_s[j * 4 + 3];
    float areaB = (bx2 - bx1) * (by2 - by1);
    float iw = fminf(ax2, bx2) - fmaxf(ax1, bx1);
    float ih = fminf(ay2, by2) - fmaxf(ay1, by1);
    float inter = fmaxf(iw, 0.0f) * fmaxf(ih, 0.0f);
    float v = inter / (areaA + areaB - inter);
    unsigned word = (unsigned)__ballot(v > IOU_T);
    if (lane == 0) iou[row * 32 + c] = word;
  }
}

// ---------------------------------------------------------------------------
// Kernel 5: greedy grouping scan (exact check-before-add / last-argmax
// semantics), bitmask incremental conflict tracking, then output.
// ---------------------------------------------------------------------------
__global__ __launch_bounds__(1024) void cwsl_kernel(
    const float* __restrict__ scores_s, const unsigned* __restrict__ iou,
    const int* __restrict__ counts, unsigned* __restrict__ recsel,
    float* __restrict__ out) {
  __shared__ float rv[N_BOX];
  __shared__ int   ri[N_BOX];
  __shared__ int   bestIdx, anyRec;

  const int i = threadIdx.x;
  unsigned sel[32], conf[32];
  for (int w = 0; w < 32; ++w) { sel[w] = 0u; conf[w] = iou[i * 32 + w]; }
  sel[i >> 5] = 1u << (i & 31);

  int   size = 1;
  float sc   = scores_s[i];
  bool  rec  = false;
  float recsc = NEG_INF;
  const int K = counts[0];

  for (int j = i + 1; j < N_BOX; ++j) {
    if (size == K) {  // record at TOP of iteration, before any add (== do_rec)
      rec = true;
      recsc = sc;
      for (int w = 0; w < 32; ++w) recsel[i * 32 + w] = sel[w];
      break;
    }
    bool compat = ((conf[j >> 5] >> (j & 31)) & 1u) == 0u;
    if (compat && size < K) {
      sel[j >> 5] |= (1u << (j & 31));
      size += 1;
      sc += scores_s[j];
      for (int w = 0; w < 32; ++w) conf[w] |= iou[j * 32 + w];
    }
  }

  rv[i] = rec ? recsc : NEG_INF;
  ri[i] = i;
  __syncthreads();
  for (int s = 512; s > 0; s >>= 1) {
    if (i < s) {
      float v1 = rv[i], v2 = rv[i + s];
      int i1 = ri[i], i2 = ri[i + s];
      if ((v2 > v1) || (v2 == v1 && i2 > i1)) { rv[i] = v2; ri[i] = i2; }  // last argmax
    }
    __syncthreads();
  }
  if (i == 0) {
    bestIdx = ri[0];
    anyRec = (rv[0] > -1e29f) ? 1 : 0;
  }
  __syncthreads();
  unsigned bit = (recsel[bestIdx * 32 + (i >> 5)] >> (i & 31)) & 1u;
  out[i] = (anyRec && bit) ? scores_s[i] : 0.0f;
}

// ---------------------------------------------------------------------------
extern "C" void kernel_launch(void* const* d_in, const int* in_sizes, int n_in,
                              void* d_out, int out_size, void* d_ws, size_t ws_size,
                              hipStream_t stream) {
  (void)in_sizes; (void)n_in; (void)out_size; (void)ws_size;
  const float* bbox   = (const float*)d_in[0];
  const float* obj    = (const float*)d_in[1];
  const float* probs  = (const float*)d_in[2];
  const int*   counts = (const int*)d_in[3];
  float* out = (float*)d_out;

  float* partial  = (float*)d_ws;                 // 256*1024 f32
  float* score    = partial + 256 * N_BOX;        // 1024
  float* scores_s = score + N_BOX;                // 1024
  float* bbox_s   = scores_s + N_BOX;             // 4096
  int*   order    = (int*)(bbox_s + 4 * N_BOX);   // 1024
  unsigned* iou    = (unsigned*)(order + N_BOX);  // 1024*32
  unsigned* recsel = iou + N_BOX * 32;            // 1024*32

  boxsum_gemm<<<dim3(W_IMG / 16), dim3(1024), 0, stream>>>(bbox, probs, partial);
  finalize_scores<<<dim3(4), dim3(256), 0, stream>>>(bbox, obj, partial, score);
  sort_scores<<<dim3(1), dim3(1024), 0, stream>>>(bbox, score, scores_s, bbox_s, order);
  iou_bits<<<dim3(128), dim3(256), 0, stream>>>(bbox_s, iou);
  cwsl_kernel<<<dim3(1), dim3(1024), 0, stream>>>(scores_s, iou, counts, recsel, out);
}
